// ContrastiveLoss_20693152432200
// MI455X (gfx1250) — compile-verified, matrix operand order checked by reading, not run on hardware
//
#include <hip/hip_runtime.h>
#include <hip/hip_bf16.h>

#define DIM 128
#define TAU_INV_2 1.0f   // 1/(2*TAU) with TAU=0.5
#define EPSV 1e-8f
#define CHUNK 64         // B columns staged per LDS chunk
#define ROWPAD 136       // halves per LDS row: 128 + 8 pad (breaks bank conflicts)
#define MROWS 32         // rows per wave M-tile (2 WMMA accumulator tiles)

typedef __attribute__((ext_vector_type(16))) _Float16 v16h;
typedef __attribute__((ext_vector_type(8)))  _Float16 v8h;
typedef __attribute__((ext_vector_type(4)))  _Float16 v4h;
typedef __attribute__((ext_vector_type(8)))  float    v8f;

// Assemble a v16h WMMA fragment from two 16-byte chunks at p and p+16 halves.
// Matches the CDNA5 16-bit A/B VGPR striping: halves [kh, kh+8) in VGPR0-3 and
// [kh+16, kh+24) in VGPR4-7 (kh carries the +8 offset for lanes 16-31).
__device__ __forceinline__ v16h frag_from(const _Float16* p) {
  v8h lo = *(const v8h*)(p);
  v8h hi = *(const v8h*)(p + 16);
  return __builtin_shufflevector(lo, hi, 0, 1, 2, 3, 4, 5, 6, 7,
                                         8, 9, 10, 11, 12, 13, 14, 15);
}

// Fragment straight from a row-major [N x 128] f16 global matrix.
__device__ __forceinline__ v16h load_frag_g(const _Float16* __restrict__ base,
                                            int row0, int k0, int lane) {
  const int r  = row0 + (lane & 15);
  const int kh = k0 + ((lane >> 4) << 3);
  return frag_from(base + (size_t)r * DIM + kh);
}

__device__ __forceinline__ void wait_asynccnt0() {
#if __has_builtin(__builtin_amdgcn_s_wait_asynccnt)
  __builtin_amdgcn_s_wait_asynccnt(0);
#else
  asm volatile("s_wait_asynccnt 0x0" ::: "memory");
#endif
}

// ---------------------------------------------------------------------------
// Kernel 1: per-row prep. One wave (32 lanes) per row of 128 floats.
//  - convert Tx1/Tx2 -> f16 workspace copies (WMMA operands)
//  - sq1[i] = ||Tx1_i||^2, sq2[i] = ||Tx2_i||^2 (exact fp32)
//  - attr[i] = exp(-||Tx1_i - Tx2_i||^2)
// ---------------------------------------------------------------------------
__global__ void __launch_bounds__(256)
prep_kernel(const float* __restrict__ Tx1, const float* __restrict__ Tx2,
            unsigned short* __restrict__ Ah_u, unsigned short* __restrict__ Bh_u,
            float* __restrict__ sq1, float* __restrict__ sq2,
            float* __restrict__ attr) {
  _Float16* Ah = (_Float16*)Ah_u;
  _Float16* Bh = (_Float16*)Bh_u;
  const int lane = threadIdx.x & 31;
  const int row  = blockIdx.x * (blockDim.x >> 5) + (threadIdx.x >> 5);

  const float4 a = *((const float4*)(Tx1 + (size_t)row * DIM) + lane);
  const float4 b = *((const float4*)(Tx2 + (size_t)row * DIM) + lane);

  float s1 = a.x * a.x + a.y * a.y + a.z * a.z + a.w * a.w;
  float s2 = b.x * b.x + b.y * b.y + b.z * b.z + b.w * b.w;
  float dx = a.x - b.x, dy = a.y - b.y, dz = a.z - b.z, dw = a.w - b.w;
  float dd = dx * dx + dy * dy + dz * dz + dw * dw;

  v4h ah = { (_Float16)a.x, (_Float16)a.y, (_Float16)a.z, (_Float16)a.w };
  v4h bh = { (_Float16)b.x, (_Float16)b.y, (_Float16)b.z, (_Float16)b.w };
  *(v4h*)(Ah + (size_t)row * DIM + lane * 4) = ah;
  *(v4h*)(Bh + (size_t)row * DIM + lane * 4) = bh;

  #pragma unroll
  for (int off = 16; off > 0; off >>= 1) {
    s1 += __shfl_xor(s1, off, 32);
    s2 += __shfl_xor(s2, off, 32);
    dd += __shfl_xor(dd, off, 32);
  }
  if (lane == 0) {
    sq1[row]  = s1;
    sq2[row]  = s2;
    attr[row] = __expf(-dd * TAU_INV_2);
  }
}

// ---------------------------------------------------------------------------
// Kernel 2: WMMA cross kernel, async global->LDS B staging, 32-row M-blocking.
//   - 8 waves/WG, each wave owns a 32-row M-tile: 8 resident A fragments so
//     every staged B fragment feeds TWO accumulator tiles (8 wmma : 8 ds_load)
//   - B staged in 64-column double-buffered LDS chunks via
//     global_load_async_to_lds_b128 (ASYNCcnt), overlapped with compute
//   - fragments fed from LDS (ds_load_b128, bank-conflict-free via row pad)
//   - fused exp epilogue into per-lane row-sum partials; one 16-lane shuffle
//     reduction at the very end; no atomics (each row owned by one wave)
// ---------------------------------------------------------------------------
__global__ void __launch_bounds__(256)
cross_kernel(const unsigned short* __restrict__ Ah_u,
             const unsigned short* __restrict__ Bh_u,
             const float* __restrict__ sq1, const float* __restrict__ sq2,
             float* __restrict__ repsum, int N) {
  __shared__ __align__(16) _Float16 Bs[2][CHUNK][ROWPAD];

  const _Float16* Ah = (const _Float16*)Ah_u;
  const _Float16* Bh = (const _Float16*)Bh_u;
  const int tid   = threadIdx.x;
  const int lane  = tid & 31;
  const int mtile = blockIdx.x * (blockDim.x >> 5) + (tid >> 5);
  const int i0    = mtile * MROWS;
  const int khalf = (lane >> 4) << 3;

  // A fragments: 32 rows x full K=128, resident for the whole kernel (64 VGPRs)
  v16h af[2][4];
  #pragma unroll
  for (int h = 0; h < 2; ++h) {
    #pragma unroll
    for (int k = 0; k < 4; ++k) af[h][k] = load_frag_g(Ah, i0 + h * 16, k * 32, lane);
  }

  // C/D layout: element v -> row m = v + 8*(lane>=16), col n = lane&15
  const int mb[2] = { i0 + khalf, i0 + 16 + khalf };
  float s1v[2][8];
  float s[2][8];
  #pragma unroll
  for (int h = 0; h < 2; ++h) {
    #pragma unroll
    for (int v = 0; v < 8; ++v) { s1v[h][v] = sq1[mb[h] + v]; s[h][v] = 0.0f; }
  }

  // Issue the async copy of one 64-col B chunk into Bs[buf].
  // 64 rows x 256B = 1024 16-byte pieces; 256 threads -> 4 pieces each.
  auto stage = [&](int jc, int buf) {
    const _Float16* src = Bh + (size_t)jc * CHUNK * DIM;
    #pragma unroll
    for (int i = 0; i < (CHUNK * 16) / 256; ++i) {
      const int p   = i * 256 + tid;
      const int row = p >> 4;         // 16 pieces per 128-half row
      const int col = p & 15;
      const unsigned ldsb = (unsigned)(size_t)(&Bs[buf][row][col * 8]);
      const void* g = (const void*)(src + (size_t)p * 8);
      asm volatile("global_load_async_to_lds_b128 %0, %1, off"
                   :: "v"(ldsb), "v"(g) : "memory");
    }
  };

  const int nchunks = N / CHUNK;
  stage(0, 0);
  wait_asynccnt0();
  __syncthreads();

  for (int jc = 0; jc < nchunks; ++jc) {
    const int cur = jc & 1;
    if (jc + 1 < nchunks) stage(jc + 1, cur ^ 1);  // prefetch next chunk

    #pragma unroll
    for (int sub = 0; sub < CHUNK / 16; ++sub) {
      const int jg = jc * CHUNK + sub * 16;
      const float s2n = sq2[jg + (lane & 15)];
      const _Float16* bp = &Bs[cur][sub * 16 + (lane & 15)][khalf];

      const v16h b0 = frag_from(bp + 0);
      const v16h b1 = frag_from(bp + 32);
      const v16h b2 = frag_from(bp + 64);
      const v16h b3 = frag_from(bp + 96);

      v8f c0 = {}, c1 = {};
      c0 = __builtin_amdgcn_wmma_f32_16x16x32_f16(false, af[0][0], false, b0, (short)0, c0, false, false);
      c1 = __builtin_amdgcn_wmma_f32_16x16x32_f16(false, af[1][0], false, b0, (short)0, c1, false, false);
      c0 = __builtin_amdgcn_wmma_f32_16x16x32_f16(false, af[0][1], false, b1, (short)0, c0, false, false);
      c1 = __builtin_amdgcn_wmma_f32_16x16x32_f16(false, af[1][1], false, b1, (short)0, c1, false, false);
      c0 = __builtin_amdgcn_wmma_f32_16x16x32_f16(false, af[0][2], false, b2, (short)0, c0, false, false);
      c1 = __builtin_amdgcn_wmma_f32_16x16x32_f16(false, af[1][2], false, b2, (short)0, c1, false, false);
      c0 = __builtin_amdgcn_wmma_f32_16x16x32_f16(false, af[0][3], false, b3, (short)0, c0, false, false);
      c1 = __builtin_amdgcn_wmma_f32_16x16x32_f16(false, af[1][3], false, b3, (short)0, c1, false, false);

      #pragma unroll
      for (int v = 0; v < 8; ++v) {
        const float d0 = fmaxf(s1v[0][v] + s2n - 2.0f * c0[v], 0.0f);
        const float d1 = fmaxf(s1v[1][v] + s2n - 2.0f * c1[v], 0.0f);
        s[0][v] += __expf(-d0 * TAU_INV_2);
        s[1][v] += __expf(-d1 * TAU_INV_2);
      }
    }

    wait_asynccnt0();   // own async loads into the other buffer done
    __syncthreads();    // all waves done reading cur / writing next
  }

  // Row sums: reduce over the 16 lanes sharing the same row group.
  #pragma unroll
  for (int h = 0; h < 2; ++h) {
    #pragma unroll
    for (int v = 0; v < 8; ++v) {
      #pragma unroll
      for (int off = 1; off < 16; off <<= 1) s[h][v] += __shfl_xor(s[h][v], off, 16);
    }
  }
  if ((lane & 15) == 0) {
    #pragma unroll
    for (int h = 0; h < 2; ++h) {
      #pragma unroll
      for (int v = 0; v < 8; ++v) repsum[mb[h] + v] = s[h][v];
    }
  }
}

// ---------------------------------------------------------------------------
// Kernel 3: final reduction -> d_out = {loss, mean(attr), mean(rep)}
// ---------------------------------------------------------------------------
__global__ void __launch_bounds__(256)
finalize_kernel(const float* __restrict__ attr, const float* __restrict__ repsum,
                float* __restrict__ out, int N) {
  __shared__ float red[3][8];
  const int tid  = threadIdx.x;
  const int lane = tid & 31;
  const int wave = tid >> 5;
  const float invN = 1.0f / (float)N;

  float tl = 0.0f, ta = 0.0f, tr = 0.0f;
  for (int i = tid; i < N; i += blockDim.x) {
    const float rep = repsum[i] * invN;   // mean over j
    const float a   = attr[i];
    tl += log1pf(rep / (a + EPSV));
    ta += a;
    tr += rep;
  }
  #pragma unroll
  for (int off = 16; off > 0; off >>= 1) {
    tl += __shfl_xor(tl, off, 32);
    ta += __shfl_xor(ta, off, 32);
    tr += __shfl_xor(tr, off, 32);
  }
  if (lane == 0) { red[0][wave] = tl; red[1][wave] = ta; red[2][wave] = tr; }
  __syncthreads();
  if (wave == 0) {
    float a0 = (lane < 8) ? red[0][lane] : 0.0f;
    float a1 = (lane < 8) ? red[1][lane] : 0.0f;
    float a2 = (lane < 8) ? red[2][lane] : 0.0f;
    #pragma unroll
    for (int off = 4; off > 0; off >>= 1) {
      a0 += __shfl_xor(a0, off, 32);
      a1 += __shfl_xor(a1, off, 32);
      a2 += __shfl_xor(a2, off, 32);
    }
    if (lane == 0) {
      out[0] = a0 * invN;   // loss
      out[1] = a1 * invN;   // mean(attraction)
      out[2] = a2 * invN;   // mean(repulsion)
    }
  }
}

extern "C" void kernel_launch(void* const* d_in, const int* in_sizes, int n_in,
                              void* d_out, int out_size, void* d_ws, size_t ws_size,
                              hipStream_t stream) {
  const float* Tx1 = (const float*)d_in[0];
  const float* Tx2 = (const float*)d_in[1];
  float* out = (float*)d_out;
  const int N = in_sizes[0] / DIM;   // 16384

  char* ws = (char*)d_ws;
  const size_t halfBytes = (size_t)N * DIM * sizeof(unsigned short);
  unsigned short* Ah  = (unsigned short*)(ws);
  unsigned short* Bh  = (unsigned short*)(ws + halfBytes);
  float* sq1    = (float*)(ws + 2 * halfBytes);
  float* sq2    = sq1 + N;
  float* attr   = sq2 + N;
  float* repsum = attr + N;

  // 1) prep: f16 copies + row norms + attraction (one wave per row)
  prep_kernel<<<dim3(N / 8), dim3(256), 0, stream>>>(Tx1, Tx2, Ah, Bh, sq1, sq2, attr);

  // 2) WMMA cross-distance + async-LDS B staging + fused exp row sums
  //    one wave per 32-row tile -> N/32 waves -> /8 waves per block
  cross_kernel<<<dim3(N / MROWS / 8), dim3(256), 0, stream>>>(Ah, Bh, sq1, sq2, repsum, N);

  // 3) final scalar reduction
  finalize_kernel<<<dim3(1), dim3(256), 0, stream>>>(attr, repsum, out, N);
}
